// DeepManualLSTM_6914897346634
// MI455X (gfx1250) — compile-verified, matrix operand order checked by reading, not run on hardware
//
#include <hip/hip_runtime.h>
#include <hip/hip_bf16.h>
#include <math.h>

// ---------------------------------------------------------------------------
// DeepManualLSTM for MI455X (gfx1250): bf16 WMMA fused-gate LSTM steps.
// ---------------------------------------------------------------------------

typedef __bf16 bf16_t;
typedef __attribute__((ext_vector_type(16))) __bf16 v16bf;
typedef __attribute__((ext_vector_type(8)))  __bf16 v8bf;
typedef __attribute__((ext_vector_type(8)))  float  v8f;

#define BSZ    1024
#define TSTEPS 48
#define INDIM  64
#define HDIM   512
#define GDIM   2048   // 4*H

union FragU { v16bf v; v8bf h[2]; };

// A fragment: 16x32 bf16 tile from a row-major [rows x stride] bf16 matrix.
// Per ISA 16-bit A layout: lanes 0-15 hold row M=lane, K = {0..7, 16..23};
// lanes 16-31 hold row M=lane-16, K = {8..15, 24..31}.
__device__ __forceinline__ v16bf load_a_frag(const bf16_t* __restrict__ A,
                                             int stride, int m0, int k0,
                                             int lane) {
  int row  = m0 + (lane & 15);
  int koff = (lane >> 4) << 3;  // 0 or 8
  const bf16_t* p = A + (size_t)row * (size_t)stride + (size_t)(k0 + koff);
  FragU f;
  f.h[0] = *reinterpret_cast<const v8bf*>(p);        // K  = k0+koff   .. +7
  f.h[1] = *reinterpret_cast<const v8bf*>(p + 16);   // K  = k0+koff+16.. +23
  return f.v;
}

// B fragment: pre-packed so each lane reads 16 contiguous bf16 (32B).
__device__ __forceinline__ v16bf load_b_frag(const bf16_t* __restrict__ Wp,
                                             size_t fragIdx, int lane) {
  const bf16_t* p = Wp + (fragIdx * 32 + (size_t)lane) * 16;
  FragU f;
  f.h[0] = *reinterpret_cast<const v8bf*>(p);
  f.h[1] = *reinterpret_cast<const v8bf*>(p + 8);
  return f.v;
}

__device__ __forceinline__ float sigmoidf_(float x) {
  return 1.0f / (1.0f + __expf(-x));
}

// ---------------------------------------------------------------------------
// Fused LSTM step: g = [Ax | Ah] @ Wpacked + b; gates; C,h update.
// Grid: (BSZ/128, HDIM/32), block 256 (8 waves).
// Wave w: gate = w>>1, row-half = w&1. Each wave: 4 Mtiles x 2 Ntiles WMMA.
// ---------------------------------------------------------------------------
__global__ __launch_bounds__(256)
void lstm_step_kernel(const bf16_t* __restrict__ Ax, int ax_stride, int Kx,
                      const bf16_t* __restrict__ Ah,          // [BSZ x HDIM]
                      const bf16_t* __restrict__ Wp,          // packed bf16
                      int nkc_total,                          // (Kx+HDIM)/32
                      const float* __restrict__ bias,         // [4H]
                      float* __restrict__ Cst,                // [BSZ x HDIM]
                      bf16_t* __restrict__ Hout)              // [BSZ x HDIM]
{
  __shared__ float lds[4 * 128 * 32];   // 64 KB: gates x rows x cols

  const int tid   = threadIdx.x;
  const int lane  = tid & 31;
  const int wave  = tid >> 5;
  const int gate  = wave >> 1;          // 0..3 (f,i,c,o)
  const int mhalf = wave & 1;
  const int mblk  = blockIdx.x * 128;
  const int m0    = mblk + mhalf * 64;
  const int nh0   = blockIdx.y * 32;                    // hidden-col base
  const int nt0   = (gate * HDIM + nh0) >> 4;           // packed n-tile index

  v8f acc[4][2] = {};

  const int nkx = Kx >> 5;
  for (int kc = 0; kc < nkx; ++kc) {
    v16bf a[4];
#pragma unroll
    for (int mi = 0; mi < 4; ++mi)
      a[mi] = load_a_frag(Ax, ax_stride, m0 + mi * 16, kc * 32, lane);
#pragma unroll
    for (int ni = 0; ni < 2; ++ni) {
      v16bf b = load_b_frag(Wp, (size_t)(nt0 + ni) * nkc_total + kc, lane);
#pragma unroll
      for (int mi = 0; mi < 4; ++mi)
        acc[mi][ni] = __builtin_amdgcn_wmma_f32_16x16x32_bf16(
            false, a[mi], false, b, (short)0, acc[mi][ni], false, false);
    }
  }

  const int nkh = HDIM >> 5;  // 16
  for (int kc = 0; kc < nkh; ++kc) {
    v16bf a[4];
#pragma unroll
    for (int mi = 0; mi < 4; ++mi)
      a[mi] = load_a_frag(Ah, HDIM, m0 + mi * 16, kc * 32, lane);
#pragma unroll
    for (int ni = 0; ni < 2; ++ni) {
      v16bf b = load_b_frag(
          Wp, (size_t)(nt0 + ni) * nkc_total + (size_t)(nkx + kc), lane);
#pragma unroll
      for (int mi = 0; mi < 4; ++mi)
        acc[mi][ni] = __builtin_amdgcn_wmma_f32_16x16x32_bf16(
            false, a[mi], false, b, (short)0, acc[mi][ni], false, false);
    }
  }

  // Spill accumulators to LDS in (gate, row, col) order.
  // C/D layout: lanes 0-15: N=lane, M=r; lanes 16-31: N=lane-16, M=8+r.
#pragma unroll
  for (int mi = 0; mi < 4; ++mi) {
#pragma unroll
    for (int ni = 0; ni < 2; ++ni) {
#pragma unroll
      for (int r = 0; r < 8; ++r) {
        int lm = mhalf * 64 + mi * 16 + ((lane >> 4) << 3) + r;
        int ln = ni * 16 + (lane & 15);
        lds[(gate * 128 + lm) * 32 + ln] = acc[mi][ni][r];
      }
    }
  }
  __syncthreads();

  // Gating epilogue: 128x32 elements over 256 threads.
  for (int idx = tid; idx < 128 * 32; idx += 256) {
    int lm  = idx >> 5;
    int ln  = idx & 31;
    int row = mblk + lm;
    int col = nh0 + ln;
    float gf = lds[(0 * 128 + lm) * 32 + ln] + bias[0 * HDIM + col];
    float gi = lds[(1 * 128 + lm) * 32 + ln] + bias[1 * HDIM + col];
    float gc = lds[(2 * 128 + lm) * 32 + ln] + bias[2 * HDIM + col];
    float go = lds[(3 * 128 + lm) * 32 + ln] + bias[3 * HDIM + col];
    float f = sigmoidf_(gf);
    float i = sigmoidf_(gi);
    float c = tanhf(gc);
    float o = sigmoidf_(go);
    size_t sidx = (size_t)row * HDIM + col;
    float Cn = f * Cst[sidx] + i * c;
    Cst[sidx]  = Cn;
    Hout[sidx] = (bf16_t)(o * tanhf(Cn));
  }
}

// ---------------------------------------------------------------------------
// Weight repack: fp32 [Wx(Kx x 4H); Wh(Kh x 4H)] -> bf16 WMMA-B fragments.
// packed[((nt*nkc + kc)*32 + lane)*16 + j], j/lane mapping per 16-bit B layout.
// ---------------------------------------------------------------------------
__global__ __launch_bounds__(256)
void pack_w_kernel(const float* __restrict__ Wx, int Kx,
                   const float* __restrict__ Wh, int Kh,
                   bf16_t* __restrict__ out)
{
  const int nkc = (Kx + Kh) >> 5;
  const size_t total = (size_t)(GDIM / 16) * (size_t)nkc * 512;
  for (size_t idx = (size_t)blockIdx.x * blockDim.x + threadIdx.x;
       idx < total; idx += (size_t)gridDim.x * blockDim.x) {
    int    j    = (int)(idx & 15);
    int    lane = (int)((idx >> 4) & 31);
    size_t frag = idx >> 9;
    int    kc   = (int)(frag % (size_t)nkc);
    int    nt   = (int)(frag / (size_t)nkc);
    int    n    = nt * 16 + (lane & 15);
    int    kh   = (lane >> 4) << 3;                 // 0 or 8
    int    k    = kc * 32 + ((j < 8) ? (j + kh) : (16 + (j - 8) + kh));
    float  v    = (k < Kx) ? Wx[(size_t)k * GDIM + n]
                           : Wh[(size_t)(k - Kx) * GDIM + n];
    out[idx] = (bf16_t)v;
  }
}

__global__ __launch_bounds__(256)
void f32_to_bf16_kernel(const float* __restrict__ in, bf16_t* __restrict__ out,
                        size_t n)
{
  for (size_t i = (size_t)blockIdx.x * blockDim.x + threadIdx.x; i < n;
       i += (size_t)gridDim.x * blockDim.x)
    out[i] = (bf16_t)in[i];
}

// ---------------------------------------------------------------------------
// Final FC: out[b] = sum_k h[b][k] * w[k] + b0   (OUT == 1)
// ---------------------------------------------------------------------------
__global__ __launch_bounds__(256)
void fc_kernel(const bf16_t* __restrict__ h, const float* __restrict__ w,
               const float* __restrict__ b, float* __restrict__ out)
{
  __shared__ float red[256];
  int row = blockIdx.x;
  float s = 0.0f;
  for (int k = threadIdx.x; k < HDIM; k += 256)
    s += (float)h[(size_t)row * HDIM + k] * w[k];
  red[threadIdx.x] = s;
  __syncthreads();
  for (int off = 128; off > 0; off >>= 1) {
    if (threadIdx.x < off) red[threadIdx.x] += red[threadIdx.x + off];
    __syncthreads();
  }
  if (threadIdx.x == 0) out[row] = red[0] + b[0];
}

// ---------------------------------------------------------------------------
extern "C" void kernel_launch(void* const* d_in, const int* in_sizes, int n_in,
                              void* d_out, int out_size, void* d_ws,
                              size_t ws_size, hipStream_t stream) {
  (void)in_sizes; (void)n_in; (void)out_size; (void)ws_size;

  const float* x    = (const float*)d_in[0];
  const float* Wx0  = (const float*)d_in[1];
  const float* Wh0  = (const float*)d_in[2];
  const float* b0   = (const float*)d_in[3];
  const float* Wx1  = (const float*)d_in[4];
  const float* Wh1  = (const float*)d_in[5];
  const float* b1   = (const float*)d_in[6];
  const float* Wx2  = (const float*)d_in[7];
  const float* Wh2  = (const float*)d_in[8];
  const float* b2   = (const float*)d_in[9];
  const float* fcw  = (const float*)d_in[10];
  const float* fcb  = (const float*)d_in[11];
  float*       out  = (float*)d_out;

  // ---- workspace layout (all sizes are 256B-aligned naturally) ----
  char* ws = (char*)d_ws;
  size_t off = 0;
  const size_t xb_elems  = (size_t)BSZ * TSTEPS * INDIM;
  bf16_t* xb  = (bf16_t*)(ws + off); off += xb_elems * sizeof(bf16_t);
  const int nkc0 = (INDIM + HDIM) / 32;   // 18
  const int nkc1 = (HDIM + HDIM) / 32;    // 32
  const size_t w0_elems = (size_t)(GDIM / 16) * nkc0 * 512;
  const size_t w1_elems = (size_t)(GDIM / 16) * nkc1 * 512;
  bf16_t* w0p = (bf16_t*)(ws + off); off += w0_elems * sizeof(bf16_t);
  bf16_t* w1p = (bf16_t*)(ws + off); off += w1_elems * sizeof(bf16_t);
  bf16_t* w2p = (bf16_t*)(ws + off); off += w1_elems * sizeof(bf16_t);
  const size_t BH = (size_t)BSZ * HDIM;
  bf16_t* hb  = (bf16_t*)(ws + off); off += 2 * 3 * BH * sizeof(bf16_t);
  float*  cst = (float*)(ws + off);  off += 3 * BH * sizeof(float);

  // ---- zero init states (deterministic per call) ----
  hipMemsetAsync(hb,  0, 2 * 3 * BH * sizeof(bf16_t), stream);
  hipMemsetAsync(cst, 0, 3 * BH * sizeof(float), stream);

  // ---- prep: x -> bf16, weights -> packed bf16 fragments ----
  f32_to_bf16_kernel<<<2048, 256, 0, stream>>>(x, xb, xb_elems);
  pack_w_kernel<<<2048, 256, 0, stream>>>(Wx0, INDIM, Wh0, HDIM, w0p);
  pack_w_kernel<<<2048, 256, 0, stream>>>(Wx1, HDIM,  Wh1, HDIM, w1p);
  pack_w_kernel<<<2048, 256, 0, stream>>>(Wx2, HDIM,  Wh2, HDIM, w2p);

  // ---- recurrence: 48 steps x 3 layers, ping-pong h buffers ----
  dim3 gg(BSZ / 128, HDIM / 32);   // (8, 16)
  for (int t = 0; t < TSTEPS; ++t) {
    int p = t & 1;
    bf16_t* hprev = hb + (size_t)p * 3 * BH;
    bf16_t* hnew  = hb + (size_t)(1 - p) * 3 * BH;
    // layer 0: x_t (stride T*IN) + h0_prev
    lstm_step_kernel<<<gg, 256, 0, stream>>>(
        xb + (size_t)t * INDIM, TSTEPS * INDIM, INDIM,
        hprev + 0 * BH, w0p, nkc0, b0, cst + 0 * BH, hnew + 0 * BH);
    // layer 1: h0_new + h1_prev
    lstm_step_kernel<<<gg, 256, 0, stream>>>(
        hnew + 0 * BH, HDIM, HDIM,
        hprev + 1 * BH, w1p, nkc1, b1, cst + 1 * BH, hnew + 1 * BH);
    // layer 2: h1_new + h2_prev
    lstm_step_kernel<<<gg, 256, 0, stream>>>(
        hnew + 1 * BH, HDIM, HDIM,
        hprev + 2 * BH, w2p, nkc1, b2, cst + 2 * BH, hnew + 2 * BH);
  }

  // after t=47 (odd), final h lives in parity-0 buffer; take layer 2.
  fc_kernel<<<BSZ, 256, 0, stream>>>(hb + 2 * BH, fcw, fcb, out);
}
